// TransformerEncoderLayer_22170621182747
// MI455X (gfx1250) — matrix operand order repaired and checked by reading, not compiled, not run on hardware
//
#include <hip/hip_runtime.h>
#include <math.h>

// ---------------------------------------------------------------------------
// Transformer encoder layer, CDNA5 (gfx1250), wave32 WMMA f16->f32.
// B=8, S=1024, D=1024, H=16, HD=64, FF=4096
// ---------------------------------------------------------------------------
#define BATCH 8
#define SEQ   1024
#define DIM   1024
#define HEADS 16
#define HDIM  64
#define FFD   4096
#define BS    (BATCH * SEQ)     // 8192 rows

typedef _Float16 v4h  __attribute__((ext_vector_type(4)));
typedef _Float16 v8h  __attribute__((ext_vector_type(8)));
typedef _Float16 v16h __attribute__((ext_vector_type(16)));
typedef float    v8f  __attribute__((ext_vector_type(8)));

static __device__ __forceinline__ v8f wmma_f16(v16h a, v16h b, v8f c) {
  // v_wmma_f32_16x16x32_f16
  return __builtin_amdgcn_wmma_f32_16x16x32_f16(false, a, false, b,
                                                (short)0, c, false, false);
}

// Two contiguous 16B (8-half) LDS chunks -> one 16-half fragment register set.
static __device__ __forceinline__ v16h frag2(const _Float16* p0, const _Float16* p1) {
  v8h a = *(const v8h*)p0;
  v8h b = *(const v8h*)p1;
  return __builtin_shufflevector(a, b, 0,1,2,3,4,5,6,7,8,9,10,11,12,13,14,15);
}

// 8 contiguous f32 from global -> 8 halfs
static __device__ __forceinline__ v8h cvt8(const float* p) {
  float4 x = *(const float4*)p;
  float4 y = *(const float4*)(p + 4);
  v8h r;
  r[0]=(_Float16)x.x; r[1]=(_Float16)x.y; r[2]=(_Float16)x.z; r[3]=(_Float16)x.w;
  r[4]=(_Float16)y.x; r[5]=(_Float16)y.y; r[6]=(_Float16)y.z; r[7]=(_Float16)y.w;
  return r;
}

static __device__ __forceinline__ v8f zero8() {
  v8f r;
#pragma unroll
  for (int i = 0; i < 8; ++i) r[i] = 0.0f;
  return r;
}

// ---------------------------------------------------------------------------
// Generic GEMM: C[M,N] = epi( A[M,K] @ B[K,N] + bias + residual )
// Block tile 128(M) x 128(N), K-step 32, 256 threads = 8 waves (4x2 of 32x64).
// Register-prefetch software pipeline over the K loop.
// HEADED==1: B is [H][K][64] with column n -> head n/64, within-head n%64.
// GELU==1:   exact GELU after bias, before residual.
// ---------------------------------------------------------------------------
template<int HEADED>
static __device__ __forceinline__
void gemm_load_gl(const float* __restrict__ A, const float* __restrict__ Bm,
                  int N, int K, int blk_m, int blk_n, int k0, int tid,
                  float4 aR[4], float4 bR[4]) {
  const int col4 = (tid & 7) * 4;
#pragma unroll
  for (int r = 0; r < 4; ++r) {
    const int row = (tid >> 3) + r * 32;
    aR[r] = *(const float4*)(A + (size_t)(blk_m + row) * K + k0 + col4);
  }
  const int n4 = (tid & 31) * 4;
#pragma unroll
  for (int r = 0; r < 4; ++r) {
    const int kr = (tid >> 5) + r * 8;
    const float* bp;
    if (HEADED) {
      const int ncol = blk_n + n4;
      bp = Bm + ((size_t)(ncol >> 6) * K + (k0 + kr)) * 64 + (ncol & 63);
    } else {
      bp = Bm + (size_t)(k0 + kr) * N + blk_n + n4;
    }
    bR[r] = *(const float4*)bp;
  }
}

template<int HEADED, int GELU>
__global__ __launch_bounds__(256)
void gemm_wmma_kernel(const float* __restrict__ A, const float* __restrict__ Bm,
                      const float* __restrict__ bias, const float* __restrict__ res,
                      float* __restrict__ C, int M, int N, int K) {
  __shared__ _Float16 As[128][40];   // [m][k] pitch 40 halfs (80B)
  __shared__ _Float16 Bs[128][40];   // [n][k] transposed

  const int tid    = threadIdx.x;
  const int lane   = tid & 31;
  const int w      = tid >> 5;
  const int l      = lane & 15;
  const int hi     = lane >> 4;
  const int wm     = (w & 3) * 32;
  const int wn     = (w >> 2) * 64;
  const int blk_m  = blockIdx.y * 128;
  const int blk_n  = blockIdx.x * 128;

  v8f acc[2][4];
#pragma unroll
  for (int mi = 0; mi < 2; ++mi)
#pragma unroll
    for (int ni = 0; ni < 4; ++ni) acc[mi][ni] = zero8();

  float4 aR[4], bR[4];
  gemm_load_gl<HEADED>(A, Bm, N, K, blk_m, blk_n, 0, tid, aR, bR);

  for (int k0 = 0; k0 < K; k0 += 32) {
    // ---- commit prefetched registers to LDS
    {
      const int col4 = (tid & 7) * 4;
#pragma unroll
      for (int r = 0; r < 4; ++r) {
        const int row = (tid >> 3) + r * 32;
        v4h h4;
        h4[0]=(_Float16)aR[r].x; h4[1]=(_Float16)aR[r].y;
        h4[2]=(_Float16)aR[r].z; h4[3]=(_Float16)aR[r].w;
        *(v4h*)&As[row][col4] = h4;
      }
      const int n4 = (tid & 31) * 4;
#pragma unroll
      for (int r = 0; r < 4; ++r) {
        const int kr = (tid >> 5) + r * 8;
        Bs[n4 + 0][kr] = (_Float16)bR[r].x;
        Bs[n4 + 1][kr] = (_Float16)bR[r].y;
        Bs[n4 + 2][kr] = (_Float16)bR[r].z;
        Bs[n4 + 3][kr] = (_Float16)bR[r].w;
      }
    }
    __syncthreads();

    // ---- issue next K-step's global loads (overlap with WMMA below)
    if (k0 + 32 < K)
      gemm_load_gl<HEADED>(A, Bm, N, K, blk_m, blk_n, k0 + 32, tid, aR, bR);

    // ---- fragments + 8 WMMAs per wave
    v16h af[2];
#pragma unroll
    for (int mi = 0; mi < 2; ++mi) {
      const _Float16* ar = &As[wm + mi * 16 + l][0];
      af[mi] = frag2(ar + hi * 8, ar + 16 + hi * 8);
    }
#pragma unroll
    for (int ni = 0; ni < 4; ++ni) {
      const _Float16* br = &Bs[wn + ni * 16 + l][0];
      v16h bf = frag2(br + hi * 8, br + 16 + hi * 8);
#pragma unroll
      for (int mi = 0; mi < 2; ++mi)
        acc[mi][ni] = wmma_f16(af[mi], bf, acc[mi][ni]);
    }
    __syncthreads();
  }

  // ---- epilogue
#pragma unroll
  for (int mi = 0; mi < 2; ++mi) {
#pragma unroll
    for (int ni = 0; ni < 4; ++ni) {
      const int col = blk_n + wn + ni * 16 + l;
#pragma unroll
      for (int j = 0; j < 8; ++j) {
        const int row = blk_m + wm + mi * 16 + j + hi * 8;
        float vv = acc[mi][ni][j];
        if (bias) vv += bias[col];
        if (GELU) vv = 0.5f * vv * (1.0f + erff(vv * 0.70710678118654752f));
        if (res) vv += res[(size_t)row * N + col];
        C[(size_t)row * N + col] = vv;
      }
    }
  }
}

// ---------------------------------------------------------------------------
// LayerNorm: one 256-thread block per row of D=1024.
// ---------------------------------------------------------------------------
__global__ __launch_bounds__(256)
void ln_kernel(const float* __restrict__ x, const float* __restrict__ g,
               const float* __restrict__ b, float* __restrict__ y) {
  __shared__ float red[16];
  const int tid  = threadIdx.x;
  const int lane = tid & 31;
  const int w    = tid >> 5;
  const size_t base = (size_t)blockIdx.x * DIM;

  float s = 0.0f, ss = 0.0f;
  float v[4];
#pragma unroll
  for (int i = 0; i < 4; ++i) {
    v[i] = x[base + tid + i * 256];
    s  += v[i];
    ss += v[i] * v[i];
  }
#pragma unroll
  for (int off = 16; off >= 1; off >>= 1) {
    s  += __shfl_xor(s,  off);
    ss += __shfl_xor(ss, off);
  }
  if (lane == 0) { red[w] = s; red[w + 8] = ss; }
  __syncthreads();
  if (w == 0 && lane < 2) {
    float acc = 0.0f;
#pragma unroll
    for (int i = 0; i < 8; ++i) acc += red[lane * 8 + i];
    red[lane * 8] = acc;   // red[0]=sum, red[8]=sumsq
  }
  __syncthreads();
  const float mu   = red[0] * (1.0f / DIM);
  const float var  = red[8] * (1.0f / DIM) - mu * mu;
  const float rstd = rsqrtf(var + 1e-5f);
#pragma unroll
  for (int i = 0; i < 4; ++i) {
    const int c = tid + i * 256;
    y[base + c] = (v[i] - mu) * rstd * g[c] + b[c];
  }
}

// ---------------------------------------------------------------------------
// Flash attention: block = (b, h, 64 query rows), 128 thr = 4 waves.
// Each wave: 16 query rows, full HD=64 O accumulator, online softmax.
// q/k/v layout: [B, S, D] with head h at columns [h*64, h*64+64).
// ---------------------------------------------------------------------------
__global__ __launch_bounds__(128)
void attn_kernel(const float* __restrict__ q, const float* __restrict__ k,
                 const float* __restrict__ v, float* __restrict__ ctx) {
  __shared__ _Float16 Ks[32][72];     // [key][dim]   pitch 144B
  __shared__ _Float16 Vs[64][40];     // [dim][key]   pitch 80B
  __shared__ _Float16 Ps[4][16][40];  // per-wave P tile [row][key]

  const int id  = blockIdx.x;
  const int qb  = id & 15;
  const int h   = (id >> 4) & (HEADS - 1);
  const int b   = id >> 8;

  const int tid  = threadIdx.x;
  const int w    = tid >> 5;
  const int lane = tid & 31;
  const int l    = lane & 15;
  const int hi   = lane >> 4;

  const size_t head_off = (size_t)b * SEQ * DIM + (size_t)h * HDIM;
  const float* qp = q + head_off;
  const float* kp = k + head_off;
  const float* vp = v + head_off;

  // Q fragments (16 rows x 64 dims, two K=32 steps) kept in registers.
  const int qrow = qb * 64 + w * 16 + l;
  const float* qr = qp + (size_t)qrow * DIM;
  v16h qf[2];
#pragma unroll
  for (int t = 0; t < 2; ++t) {
    v8h c0 = cvt8(qr + t * 32 + hi * 8);
    v8h c1 = cvt8(qr + t * 32 + 16 + hi * 8);
    qf[t] = __builtin_shufflevector(c0, c1, 0,1,2,3,4,5,6,7,8,9,10,11,12,13,14,15);
  }

  float m_st[8], l_st[8];
#pragma unroll
  for (int j = 0; j < 8; ++j) { m_st[j] = -1e30f; l_st[j] = 0.0f; }
  v8f o_acc[4] = {zero8(), zero8(), zero8(), zero8()};

  for (int kb = 0; kb < SEQ; kb += 32) {
    // cooperative load: K block [32][64] row-major, V block transposed [64][32]
    {
      const int d4 = (tid & 15) * 4;
#pragma unroll
      for (int r = 0; r < 4; ++r) {
        const int key = (tid >> 4) + r * 8;
        float4 kv = *(const float4*)(kp + (size_t)(kb + key) * DIM + d4);
        v4h h4;
        h4[0]=(_Float16)kv.x; h4[1]=(_Float16)kv.y;
        h4[2]=(_Float16)kv.z; h4[3]=(_Float16)kv.w;
        *(v4h*)&Ks[key][d4] = h4;
        float4 vv = *(const float4*)(vp + (size_t)(kb + key) * DIM + d4);
        Vs[d4 + 0][key] = (_Float16)vv.x;
        Vs[d4 + 1][key] = (_Float16)vv.y;
        Vs[d4 + 2][key] = (_Float16)vv.z;
        Vs[d4 + 3][key] = (_Float16)vv.w;
      }
    }
    __syncthreads();

    // ---- scores S = Q @ K^T (16x32), two 16-wide N tiles, K(dim)=64
    v8f sa[2];
#pragma unroll
    for (int ni = 0; ni < 2; ++ni) {
      const _Float16* kr = &Ks[ni * 16 + l][0];
      v16h b0 = frag2(kr + hi * 8,      kr + 16 + hi * 8);       // dims 0..31
      v16h b1 = frag2(kr + 32 + hi * 8, kr + 48 + hi * 8);  // dims 32..63
      v8f st = zero8();
      st = wmma_f16(qf[0], b0, st);
      st = wmma_f16(qf[1], b1, st);
#pragma unroll
      for (int j = 0; j < 8; ++j) st[j] *= 0.125f;  // 1/sqrt(64)
      sa[ni] = st;
    }

    // ---- online softmax (row reductions across 16-lane halves)
#pragma unroll
    for (int j = 0; j < 8; ++j) {
      float mrow = fmaxf(sa[0][j], sa[1][j]);
#pragma unroll
      for (int msk = 8; msk >= 1; msk >>= 1) mrow = fmaxf(mrow, __shfl_xor(mrow, msk));
      const float mn = fmaxf(m_st[j], mrow);
      const float al = expf(m_st[j] - mn);
      sa[0][j] = expf(sa[0][j] - mn);
      sa[1][j] = expf(sa[1][j] - mn);
      float rs = sa[0][j] + sa[1][j];
#pragma unroll
      for (int msk = 8; msk >= 1; msk >>= 1) rs += __shfl_xor(rs, msk);
      l_st[j] = l_st[j] * al + rs;
      m_st[j] = mn;
#pragma unroll
      for (int di = 0; di < 4; ++di) o_acc[di][j] *= al;
    }

    // ---- P (C layout) -> LDS -> A-fragment layout
#pragma unroll
    for (int ni = 0; ni < 2; ++ni)
#pragma unroll
      for (int j = 0; j < 8; ++j)
        Ps[w][j + hi * 8][ni * 16 + l] = (_Float16)sa[ni][j];
    asm volatile("s_wait_dscnt 0" ::: "memory");
    const _Float16* pr = &Ps[w][l][0];
    v16h pf = frag2(pr + hi * 8, pr + 16 + hi * 8);

    // ---- O += P(16x32) @ V(32x64)
#pragma unroll
    for (int di = 0; di < 4; ++di) {
      const _Float16* vr = &Vs[di * 16 + l][0];
      v16h vb = frag2(vr + hi * 8, vr + 16 + hi * 8);
      o_acc[di] = wmma_f16(pf, vb, o_acc[di]);
    }
    __syncthreads();
  }

  // ---- normalize + write ctx [B,S,D]
#pragma unroll
  for (int j = 0; j < 8; ++j) {
    const float inv = 1.0f / l_st[j];
    const int row_s = qb * 64 + w * 16 + j + hi * 8;
#pragma unroll
    for (int di = 0; di < 4; ++di) {
      const int col = h * HDIM + di * 16 + l;
      ctx[((size_t)b * SEQ + row_s) * DIM + col] = o_acc[di][j] * inv;
    }
  }
}

// ---------------------------------------------------------------------------
// Host-side orchestration.
// ---------------------------------------------------------------------------
extern "C" void kernel_launch(void* const* d_in, const int* in_sizes, int n_in,
                              void* d_out, int out_size, void* d_ws, size_t ws_size,
                              hipStream_t stream) {
  (void)in_sizes; (void)n_in; (void)out_size; (void)ws_size;

  const float* x     = (const float*)d_in[0];
  const float* ln1_g = (const float*)d_in[1];
  const float* ln1_b = (const float*)d_in[2];
  const float* ln2_g = (const float*)d_in[3];
  const float* ln2_b = (const float*)d_in[4];
  const float* Wq    = (const float*)d_in[5];
  const float* bq    = (const float*)d_in[6];
  const float* Wk    = (const float*)d_in[7];
  const float* bk    = (const float*)d_in[8];
  const float* Wv    = (const float*)d_in[9];
  const float* bv    = (const float*)d_in[10];
  const float* Wo    = (const float*)d_in[11];
  const float* bo    = (const float*)d_in[12];
  const float* W1    = (const float*)d_in[13];
  const float* b1    = (const float*)d_in[14];
  const float* W2    = (const float*)d_in[15];
  const float* b2    = (const float*)d_in[16];
  const float* Wout  = (const float*)d_in[17];
  const float* bout  = (const float*)d_in[18];
  float* out = (float*)d_out;

  float* ws = (float*)d_ws;
  const size_t N0 = (size_t)BS * DIM;          // 8.4M floats
  float* hbuf = ws;                            // hln -> h2 -> x2
  float* qbuf = ws + N0;
  float* kbuf = ws + 2 * N0;
  float* vbuf = ws + 3 * N0;
  float* cbuf = ws + 4 * N0;                   // ctx
  float* ffh  = ws + N0;                       // reuses q/k/v/ctx (4*N0 == BS*FF)
  float* x1   = ws + 5 * N0;                   // total 6*N0 floats = 192 MB

  const dim3 blk(256);
  const dim3 gD(DIM / 128, BS / 128);          // (8, 64)
  const dim3 gF(FFD / 128, BS / 128);          // (32, 64)

  // 1) h = LN1(x)
  ln_kernel<<<BS, blk, 0, stream>>>(x, ln1_g, ln1_b, hbuf);
  // 2) q,k,v = h @ W{q,k,v} + b  (headed B layout [H,D,HD])
  gemm_wmma_kernel<1,0><<<gD, blk, 0, stream>>>(hbuf, Wq, bq, nullptr, qbuf, BS, DIM, DIM);
  gemm_wmma_kernel<1,0><<<gD, blk, 0, stream>>>(hbuf, Wk, bk, nullptr, kbuf, BS, DIM, DIM);
  gemm_wmma_kernel<1,0><<<gD, blk, 0, stream>>>(hbuf, Wv, bv, nullptr, vbuf, BS, DIM, DIM);
  // 3) ctx = softmax(q k^T / 8) v  (flash, per (b,h,64 rows))
  attn_kernel<<<BATCH * HEADS * (SEQ / 64), dim3(128), 0, stream>>>(qbuf, kbuf, vbuf, cbuf);
  // 4) x1 = x + ctx @ Wo + bo
  gemm_wmma_kernel<0,0><<<gD, blk, 0, stream>>>(cbuf, Wo, bo, x, x1, BS, DIM, DIM);
  // 5) h2 = LN2(x1)
  ln_kernel<<<BS, blk, 0, stream>>>(x1, ln2_g, ln2_b, hbuf);
  // 6) ffh = gelu(h2 @ W1 + b1)
  gemm_wmma_kernel<0,1><<<gF, blk, 0, stream>>>(hbuf, W1, b1, nullptr, ffh, BS, FFD, DIM);
  // 7) x2 = x1 + ffh @ W2 + b2
  gemm_wmma_kernel<0,0><<<gD, blk, 0, stream>>>(ffh, W2, b2, x1, hbuf, BS, DIM, FFD);
  // 8) out = x2 @ Wout + bout
  gemm_wmma_kernel<0,0><<<gD, blk, 0, stream>>>(hbuf, Wout, bout, nullptr, out, BS, DIM, DIM);
}
